// SimpleTitansMemory_89489938580101
// MI455X (gfx1250) — matrix operand order repaired and checked by reading, not compiled, not run on hardware
//
#include <hip/hip_runtime.h>
#include <hip/hip_bf16.h>
#include <math.h>

// CDNA5 (gfx1250) wave32 WMMA types
typedef __attribute__((ext_vector_type(16))) __bf16 v16bf;
typedef __attribute__((ext_vector_type(8)))  float  v8f;
// TDM descriptor group types (6-arg builtin form on this toolchain)
typedef __attribute__((ext_vector_type(4))) unsigned int tdm_v4u;
typedef __attribute__((ext_vector_type(8))) int          tdm_v8i;
typedef __attribute__((ext_vector_type(4))) int          tdm_v4i;

#define TILE_N     16          // key columns per WMMA tile
#define ROW_PAD    264         // 256 + 8 bf16 elems -> 528B row stride: 16B aligned,
                               // (528/4)%64 = 4 -> banks 4m..4m+3 distinct, m=0..15
#define STEP_TILES 4           // tiles staged per TDM descriptor / barrier interval
#define CHUNK_ROWS (TILE_N * STEP_TILES)        // 64 key rows per chunk
#define CHUNK_ELEMS (CHUNK_ROWS * ROW_PAD)      // 16896 bf16 = 33 KB (padded)

#if defined(__AMDGCN__) && __has_builtin(__builtin_amdgcn_tensor_load_to_lds) && \
    __has_builtin(__builtin_amdgcn_s_wait_tensorcnt)
#define USE_TDM 1
#else
#define USE_TDM 0
#endif

// ---------------------------------------------------------------------------
// Prep: normalize keys -> bf16 kn (workspace), copy keys/vals into output
// regions (scatter pass later overwrites the selected rows).
// One block per key row; blockDim == D == 256.
// ---------------------------------------------------------------------------
__global__ __launch_bounds__(256) void titans_prep(
    const float* __restrict__ keys, const float* __restrict__ vals,
    __bf16* __restrict__ kn, float* __restrict__ outK, float* __restrict__ outV,
    int D)
{
    __shared__ float red[8];
    const int row = blockIdx.x;
    const int t   = threadIdx.x;
    const size_t off = (size_t)row * D + t;

    const float v = keys[off];
    float ss = v * v;
    #pragma unroll
    for (int m = 1; m <= 16; m <<= 1) ss += __shfl_xor(ss, m, 32);
    if ((t & 31) == 0) red[t >> 5] = ss;
    __syncthreads();
    float tot = 0.f;
    #pragma unroll
    for (int w = 0; w < 8; ++w) tot += red[w];

    const float inv = 1.0f / fmaxf(sqrtf(tot), 1e-8f);   // eps clamp as in reference
    kn[off]   = (__bf16)(v * inv);
    outK[off] = v;
    outV[off] = vals[off];
}

#if USE_TDM
// ---------------------------------------------------------------------------
// TDM: one descriptor-driven 2D copy of a CHUNK_ROWS x D bf16 chunk,
// global -> LDS, with hardware padding of 16 B after every 512 B row
// (ISA D# §8.3/8.4: pad_interval code 6 -> 512 B; pad_amount code 3 -> 16 B).
// Resulting LDS row stride = 528 B = ROW_PAD elements (bank-conflict-free).
// ---------------------------------------------------------------------------
__device__ __forceinline__ void tdm_stage_chunk(const __bf16* gsrc,
                                                const __bf16* lds_dst,
                                                int S, int D)
{
    const unsigned long long ga = (unsigned long long)(uintptr_t)gsrc;
    const unsigned int       la = (unsigned int)(uintptr_t)lds_dst; // LDS = addr[31:0]

    tdm_v4u g0;
    g0[0] = 1u;                                            // count=1 (valid user D#)
    g0[1] = la;                                            // lds_addr
    g0[2] = (unsigned int)ga;                              // global_addr[31:0]
    g0[3] = (unsigned int)((ga >> 32) & 0x01FFFFFFu)       // global_addr[56:32]
            | (2u << 30);                                  // type = 2 ("image")

    tdm_v8i g1;
    g1[0] = (int)((1u << 16)        // data_size = 1 -> 2-byte elements
            |     (1u << 20)        // pad_enable
            |     (6u << 22)        // pad_interval: 512 B
            |     (3u << 25));      // pad_amount: 16 B
    g1[1] = (int)(((unsigned)D & 0xFFFFu) << 16);   // tensor_dim0[15:0] @ bits[63:48]
    g1[2] = (int)(((unsigned)S & 0xFFFFu) << 16);   // dim0 hi=0; tensor_dim1 lo16
    g1[3] = (int)(((unsigned)D & 0xFFFFu) << 16);   // dim1 hi=0; tile_dim0 = D
    g1[4] = CHUNK_ROWS;                             // tile_dim1 = 64; tile_dim2 = 0
    g1[5] = D;                                      // tensor_dim0_stride lo32
    g1[6] = 0;
    g1[7] = 0;

    tdm_v4i g2 = {0, 0, 0, 0};                      // 2D: dims 2..4 unused
    tdm_v4i g3 = {0, 0, 0, 0};
    tdm_v8i gx = {0, 0, 0, 0, 0, 0, 0, 0};          // aux group (6-arg form), unused

    __builtin_amdgcn_tensor_load_to_lds(g0, g1, g2, g3, gx, 0);
}
#endif

// ---------------------------------------------------------------------------
// Main: per wave, 16 query rows.  Register-resident bf16 A fragments,
// double-buffered 64-row LDS key chunks (TDM-staged), 32x back-to-back
// v_wmma_f32_16x16x32_bf16 per barrier interval, running argmax, then
// gather/scatter phase.
// ---------------------------------------------------------------------------
__global__ __launch_bounds__(256) void titans_main(
    const float* __restrict__ query, const float* __restrict__ target,
    const float* __restrict__ surprise, const float* __restrict__ keys,
    const float* __restrict__ vals, const __bf16* __restrict__ kn,
    float* __restrict__ out, int B, int S, int D)
{
    __shared__ __align__(16) __bf16 tiles[2 * CHUNK_ELEMS];
    __shared__ int idxSh[8][16];

    const int t    = threadIdx.x;
    const int wave = t >> 5;
    const int lane = t & 31;
    const int m    = lane & 15;       // A-row / B-col / C-col index
    const int h    = lane >> 4;       // half-wave selector
    const int rowBase = blockIdx.x * 128 + wave * 16;

    // ---- Load 16 query rows once; normalize; keep as bf16 A fragments ----
    // 16-bit A 16x32 (contiguous, matches B): lane m (h=0) holds K = c*32+0..15,
    // lane m (h=1) holds K = c*32+16..31.
    v16bf a[8];
    float ss = 0.f;
    const float* qrowp = query + (size_t)(rowBase + m) * D;
    #pragma unroll
    for (int c = 0; c < 8; ++c) {
        const int k0 = c * 32 + h * 16;
        float fl[16];
        *(float4*)&fl[0]  = *(const float4*)(qrowp + k0);
        *(float4*)&fl[4]  = *(const float4*)(qrowp + k0 + 4);
        *(float4*)&fl[8]  = *(const float4*)(qrowp + k0 + 8);
        *(float4*)&fl[12] = *(const float4*)(qrowp + k0 + 12);
        #pragma unroll
        for (int i = 0; i < 16; ++i) { ss += fl[i] * fl[i]; a[c][i] = (__bf16)fl[i]; }
    }
    ss += __shfl_xor(ss, 16, 32);                       // combine the two half-rows
    const float inv = 1.0f / fmaxf(sqrtf(ss), 1e-8f);
    #pragma unroll
    for (int c = 0; c < 8; ++c)
        #pragma unroll
        for (int i = 0; i < 16; ++i)
            a[c][i] = (__bf16)((float)a[c][i] * inv);

#if !USE_TDM
    // Manual staging fallback: loads at loop top (into regs), store after compute.
    uint4 su[STEP_TILES][2];
    auto stageLoad = [&](int chunk) {
        #pragma unroll
        for (int i = 0; i < STEP_TILES; ++i) {
            const int srow = i * TILE_N + (t >> 4), seg = t & 15;
            const __bf16* src =
                kn + ((size_t)(chunk * CHUNK_ROWS + srow) * D + seg * 16);
            su[i][0] = *(const uint4*)(src);
            su[i][1] = *(const uint4*)(src + 8);
        }
    };
    auto stageStore = [&](int buf) {
        #pragma unroll
        for (int i = 0; i < STEP_TILES; ++i) {
            const int srow = i * TILE_N + (t >> 4), seg = t & 15;
            __bf16* dst = &tiles[buf * CHUNK_ELEMS + srow * ROW_PAD + seg * 16];
            *(uint4*)(dst)     = su[i][0];
            *(uint4*)(dst + 8) = su[i][1];
        }
    };
#endif

    float bv[8]; int bi[8];
    #pragma unroll
    for (int j = 0; j < 8; ++j) { bv[j] = -INFINITY; bi[j] = 0; }

    const int nChunks = S / CHUNK_ROWS;

    // ---- Prologue: stage chunk 0 into buffer 0 ----
#if USE_TDM
    if (wave == 0) {
        tdm_stage_chunk(kn, &tiles[0], S, D);
        __builtin_amdgcn_s_wait_tensorcnt(0);
    }
#else
    stageLoad(0);
    stageStore(0);
#endif
    __syncthreads();

    for (int ch = 0; ch < nChunks; ++ch) {
        // Kick off next-chunk staging; latency hides under 32 WMMAs.
#if USE_TDM
        if (wave == 0 && ch + 1 < nChunks)
            tdm_stage_chunk(kn + (size_t)(ch + 1) * CHUNK_ROWS * D,
                            &tiles[((ch + 1) & 1) * CHUNK_ELEMS], S, D);
#else
        if (ch + 1 < nChunks) stageLoad(ch + 1);
#endif

        const __bf16* chunkBase = &tiles[(ch & 1) * CHUNK_ELEMS];
        #pragma unroll
        for (int sub = 0; sub < STEP_TILES; ++sub) {
            // B fragment: lane holds column N=m, K = c*32 + h*16 + 0..15.
            const __bf16* tbase = chunkBase + (sub * TILE_N + m) * ROW_PAD;
            union BF { uint4 u[2]; v16bf v; };
            // Software-pipelined fragment loads: load c+1 before WMMA c so the
            // in-order DScnt allows partial waits (overlap LDS latency).
            BF cur, nxt;
            cur.u[0] = *(const uint4*)(tbase + h * 16);
            cur.u[1] = *(const uint4*)(tbase + h * 16 + 8);
            v8f acc = {};
            #pragma unroll
            for (int c = 0; c < 8; ++c) {
                if (c < 7) {
                    const __bf16* bp = tbase + (c + 1) * 32 + h * 16;
                    nxt.u[0] = *(const uint4*)(bp);
                    nxt.u[1] = *(const uint4*)(bp + 8);
                }
                acc = __builtin_amdgcn_wmma_f32_16x16x32_bf16(
                    false, a[c], false, cur.v, (short)0, acc, false, false);
                if (c < 7) cur = nxt;
            }
            // C layout: lane (m,h) holds rows M = j + h*8 at column N = m.
            const int myn = (ch * STEP_TILES + sub) * TILE_N + m;
            #pragma unroll
            for (int j = 0; j < 8; ++j)
                if (acc[j] > bv[j]) { bv[j] = acc[j]; bi[j] = myn; } // first max wins
        }

#if USE_TDM
        if (wave == 0 && ch + 1 < nChunks)
            __builtin_amdgcn_s_wait_tensorcnt(0);   // drain TDM before barrier
#else
        if (ch + 1 < nChunks) stageStore((ch + 1) & 1);
#endif
        __syncthreads();
    }

    // ---- argmax reduce across the 16-lane N groups (tie -> smaller index) ----
    #pragma unroll
    for (int j = 0; j < 8; ++j) {
        #pragma unroll
        for (int mk = 1; mk <= 8; mk <<= 1) {
            float ov = __shfl_xor(bv[j], mk, 32);
            int   oi = __shfl_xor(bi[j], mk, 32);
            if (ov > bv[j] || (ov == bv[j] && oi < bi[j])) { bv[j] = ov; bi[j] = oi; }
        }
    }
    if (m == 0) {
        #pragma unroll
        for (int j = 0; j < 8; ++j) idxSh[wave][h * 8 + j] = bi[j];
    }
    __syncthreads();

    // ---- Phase 2: gather read_out; gated scatter into new_keys/new_vals ----
    float* outRead = out;
    float* outKeys = out + (size_t)B * D;
    float* outVals = out + (size_t)(B + S) * D;
    for (int r = 0; r < 16; ++r) {
        const int idx  = idxSh[wave][r];
        const int qrow = rowBase + r;
        const float sp = surprise[qrow];
        const float g  = 0.1f / (1.0f + __expf(-4.0f * sp));   // sigmoid(4s)*LR
        const float* vrow = vals   + (size_t)idx  * D;
        const float* krow = keys   + (size_t)idx  * D;
        const float* qr   = query  + (size_t)qrow * D;
        const float* tr   = target + (size_t)qrow * D;
        float* oR = outRead + (size_t)qrow * D;
        float* oK = outKeys + (size_t)idx  * D;
        float* oV = outVals + (size_t)idx  * D;
        #pragma unroll
        for (int i = 0; i < 8; ++i) {
            const int e = lane + i * 32;           // fully coalesced
            const float vv = vrow[e];
            oR[e] = vv;                            // pre-update vals
            oK[e] = (1.f - g) * krow[e] + g * qr[e];
            oV[e] = (1.f - g) * vv      + g * tr[e];
        }
    }
}

// ---------------------------------------------------------------------------
extern "C" void kernel_launch(void* const* d_in, const int* in_sizes, int n_in,
                              void* d_out, int out_size, void* d_ws, size_t ws_size,
                              hipStream_t stream) {
    (void)n_in; (void)out_size; (void)ws_size;
    const float* query    = (const float*)d_in[0];
    const float* target   = (const float*)d_in[1];
    const float* surprise = (const float*)d_in[2];
    const float* keys     = (const float*)d_in[3];
    const float* vals     = (const float*)d_in[4];

    const int B = in_sizes[2];            // surprise is [B,1]
    const int D = in_sizes[0] / B;        // 256
    const int S = in_sizes[3] / D;        // 1024

    float*  out  = (float*)d_out;
    __bf16* kn   = (__bf16*)d_ws;                 // S*D bf16 = 512 KB scratch
    float*  outK = out + (size_t)B * D;
    float*  outV = outK + (size_t)S * D;

    titans_prep<<<S, 256, 0, stream>>>(keys, vals, kn, outK, outV, D);
    titans_main<<<B / 128, 256, 0, stream>>>(query, target, surprise, keys, vals,
                                             kn, out, B, S, D);
}
